// Block_87840671138274
// MI455X (gfx1250) — matrix operand order repaired and checked by reading, not compiled, not run on hardware
//
#include <hip/hip_runtime.h>
#include <math.h>

// ---------------------------------------------------------------------------
// Types for CDNA5 WMMA (gfx1250, wave32)
// ---------------------------------------------------------------------------
typedef __bf16 bf16_t;
typedef __attribute__((ext_vector_type(16))) __bf16 v16bf;
typedef __attribute__((ext_vector_type(8)))  __bf16 v8bf;   // 16-byte chunk
typedef __attribute__((ext_vector_type(8)))  float  v8f;

union V16U { v16bf v; v8bf h[2]; };

#define WMMA_BF16(a, b, c) \
    __builtin_amdgcn_wmma_f32_16x16x32_bf16(false, (a), false, (b), (short)0, (c), false, false)

// DPP16 cross-lane move: quad_perm xor1 = 0xB1, xor2 = 0x4E, row_ror:4 = 0x124,
// row_ror:8 = 0x128.  DPP rows are 16 lanes => stays inside each wave32 half,
// which is exactly the WMMA C-matrix lane group.
#define DPP_F32(x, ctrl) \
    __int_as_float(__builtin_amdgcn_update_dpp(0, __float_as_int(x), (ctrl), 0xf, 0xf, true))

__device__ __forceinline__ float red_max16(float x) {
    x = fmaxf(x, DPP_F32(x, 0xB1));   // pairs
    x = fmaxf(x, DPP_F32(x, 0x4E));   // quads
    x = fmaxf(x, DPP_F32(x, 0x124));  // + next quad (ror 4)
    x = fmaxf(x, DPP_F32(x, 0x128));  // + other half of row (ror 8)
    return x;
}
__device__ __forceinline__ float red_sum16(float x) {
    x += DPP_F32(x, 0xB1);
    x += DPP_F32(x, 0x4E);
    x += DPP_F32(x, 0x124);
    x += DPP_F32(x, 0x128);
    return x;
}

__device__ __forceinline__ float gelu_tanh(float x) {
    float x3 = x * x * x;
    return 0.5f * x * (1.0f + tanhf(0.7978845608028654f * (x + 0.044715f * x3)));
}

// ---------------------------------------------------------------------------
// Weight transpose + fp32 -> bf16:  Wt[n*K + k] = W[k*N + n]
// ---------------------------------------------------------------------------
__global__ void wtrans_kernel(const float* __restrict__ W, bf16_t* __restrict__ Wt,
                              int K, int N) {
    int i = blockIdx.x * 256 + threadIdx.x;
    if (i >= K * N) return;
    int n = i % N;
    int k = i / N;
    Wt[(size_t)n * K + k] = (bf16_t)W[i];
}

// ---------------------------------------------------------------------------
// LayerNorm over last dim (768) -> bf16.  One row (of 4096) per 256-thr block.
// ---------------------------------------------------------------------------
__global__ __launch_bounds__(256) void ln_bf16_kernel(const float* __restrict__ X,
                                                      const float* __restrict__ g,
                                                      const float* __restrict__ b,
                                                      bf16_t* __restrict__ out) {
    const int D = 768;
    int row = blockIdx.x;
    int t = threadIdx.x;
    const float* xr = X + (size_t)row * D;
    float vals[3];
    float s = 0.f, s2 = 0.f;
#pragma unroll
    for (int i = 0; i < 3; ++i) {
        float v = xr[t + i * 256];
        vals[i] = v;
        s += v;
        s2 += v * v;
    }
    __shared__ float sh[2][256];
    sh[0][t] = s; sh[1][t] = s2;
    __syncthreads();
    for (int off = 128; off > 0; off >>= 1) {
        if (t < off) { sh[0][t] += sh[0][t + off]; sh[1][t] += sh[1][t + off]; }
        __syncthreads();
    }
    float mean = sh[0][0] * (1.0f / D);
    float var  = sh[1][0] * (1.0f / D) - mean * mean;
    float rstd = rsqrtf(var + 1e-6f);
#pragma unroll
    for (int i = 0; i < 3; ++i) {
        int c = t + i * 256;
        out[(size_t)row * D + c] = (bf16_t)((vals[i] - mean) * rstd * g[c] + b[c]);
    }
}

// ---------------------------------------------------------------------------
// V transpose: V[B,S,H*64] bf16 -> Vt[B*H, 64, S] bf16
// ---------------------------------------------------------------------------
__global__ void vtrans_kernel(const bf16_t* __restrict__ V, bf16_t* __restrict__ Vt) {
    const int S = 1024, D = 768, NH = 12;
    int i = blockIdx.x * 256 + threadIdx.x;
    if (i >= 4096 * 768) return;
    int d  = i % D;
    int bs = i / D;
    int s  = bs % S;
    int b  = bs / S;
    int h  = d / 64;
    int hd = d % 64;
    Vt[((size_t)(b * NH + h) * 64 + hd) * S + s] = V[i];
}

// ---------------------------------------------------------------------------
// WMMA GEMM:  C[M,N] = A[M,K](bf16) * Bt[N,K](bf16)^T + bias, optional GELU.
// 256 threads = 8 waves; wave grid 4(M) x 2(N); 32x64 per wave; tile 128x128.
// 8 WMMAs per 12 b128 loads per wave K-step.
// ---------------------------------------------------------------------------
template <bool OUT_BF16, bool DO_GELU>
__global__ __launch_bounds__(256) void gemm_bf16_kernel(
    const bf16_t* __restrict__ A, const bf16_t* __restrict__ Bt,
    const float* __restrict__ bias, void* __restrict__ Cout,
    int M, int N, int K) {
    const int lane = threadIdx.x & 31;
    const int wid  = threadIdx.x >> 5;
    const int nn   = lane & 15;
    const int half = lane >> 4;
    const int wm = wid >> 1, wn = wid & 1;
    const int r0 = blockIdx.y * 128 + wm * 32;
    const int c0 = blockIdx.x * 128 + wn * 64;

    v8f acc[2][4] = {};
    for (int k = 0; k < K; k += 32) {
        // speculative prefetch of the next-next K chunk (global_prefetch_b8)
        __builtin_prefetch(A + (size_t)(r0 + nn) * K + k + 64, 0, 1);
        __builtin_prefetch(Bt + (size_t)(c0 + nn) * K + k + 64, 0, 1);
        V16U a[2], bb[4];
#pragma unroll
        for (int i = 0; i < 2; ++i) {
            // A fragment: row r0+16i+nn, 16 contiguous K at k+16h
            const bf16_t* ap = A + (size_t)(r0 + i * 16 + nn) * K + k;
            a[i].h[0] = *(const v8bf*)(ap + 16 * half);
            a[i].h[1] = *(const v8bf*)(ap + 16 * half + 8);
        }
#pragma unroll
        for (int j = 0; j < 4; ++j) {
            // B fragment: col c0+16j+nn, 16 contiguous K at k+16h
            const bf16_t* bp = Bt + (size_t)(c0 + j * 16 + nn) * K + k + 16 * half;
            bb[j].h[0] = *(const v8bf*)bp;
            bb[j].h[1] = *(const v8bf*)(bp + 8);
        }
#pragma unroll
        for (int i = 0; i < 2; ++i)
#pragma unroll
            for (int j = 0; j < 4; ++j)
                acc[i][j] = WMMA_BF16(a[i].v, bb[j].v, acc[i][j]);
    }
#pragma unroll
    for (int i = 0; i < 2; ++i)
#pragma unroll
        for (int j = 0; j < 4; ++j) {
            int col = c0 + j * 16 + nn;
            float bv = bias ? bias[col] : 0.0f;
#pragma unroll
            for (int r = 0; r < 8; ++r) {
                int row = r0 + i * 16 + 8 * half + r;
                float v = acc[i][j][r] + bv;
                if (DO_GELU) v = gelu_tanh(v);
                if (OUT_BF16)
                    ((bf16_t*)Cout)[(size_t)row * N + col] = (bf16_t)v;
                else
                    ((float*)Cout)[(size_t)row * N + col] = v;
            }
        }
}

// ---------------------------------------------------------------------------
// Flash attention (wave32 WMMA). One wave per 16-row q tile; online softmax
// with DPP16 row reductions.  Q,K: [B,S,H*64] bf16 ; Vt: [B*H,64,S] bf16 ;
// O: [B,S,H*64] bf16.  block = 128 threads (4 waves); grid = (S/64, B*H)
// ---------------------------------------------------------------------------
__global__ __launch_bounds__(128) void attn_kernel(
    const bf16_t* __restrict__ Q, const bf16_t* __restrict__ Km,
    const bf16_t* __restrict__ Vt, bf16_t* __restrict__ O) {
    const int S = 1024, D = 768, NH = 12;
    __shared__ bf16_t ptile[4][16][40];   // per-wave P tile, 16B-aligned rows

    const int lane = threadIdx.x & 31;
    const int w    = threadIdx.x >> 5;
    const int nn   = lane & 15;
    const int half = lane >> 4;
    const int qt = blockIdx.x * 4 + w;
    const int bh = blockIdx.y;
    const int b = bh / NH, h = bh % NH;

    // Q fragments (16x64 = two 16x32 A-fragments), rows = qt*16 + nn
    const bf16_t* qp = Q + (size_t)(b * S + qt * 16 + nn) * D + h * 64;
    V16U qa0, qa1;
    qa0.h[0] = *(const v8bf*)(qp + 16 * half);
    qa0.h[1] = *(const v8bf*)(qp + 8 + 16 * half);
    qa1.h[0] = *(const v8bf*)(qp + 32 + 16 * half);
    qa1.h[1] = *(const v8bf*)(qp + 40 + 16 * half);

    float mrow[8], lrow[8];
#pragma unroll
    for (int r = 0; r < 8; ++r) { mrow[r] = -1e30f; lrow[r] = 0.f; }
    v8f accv[4] = {};
    const float scale = 0.125f;   // 1/sqrt(64)

    for (int j = 0; j < S; j += 32) {
        // ---- scores: S0 = Q x K[j..j+15]^T , S1 = Q x K[j+16..j+31]^T
        v8f sc0 = {}, sc1 = {};
        const bf16_t* kp0 = Km + (size_t)(b * S + j + nn) * D + h * 64 + 16 * half;
        const bf16_t* kp1 = kp0 + (size_t)16 * D;
        V16U kf;
        kf.h[0] = *(const v8bf*)kp0;        kf.h[1] = *(const v8bf*)(kp0 + 8);
        sc0 = WMMA_BF16(qa0.v, kf.v, sc0);
        kf.h[0] = *(const v8bf*)(kp0 + 32); kf.h[1] = *(const v8bf*)(kp0 + 40);
        sc0 = WMMA_BF16(qa1.v, kf.v, sc0);
        kf.h[0] = *(const v8bf*)kp1;        kf.h[1] = *(const v8bf*)(kp1 + 8);
        sc1 = WMMA_BF16(qa0.v, kf.v, sc1);
        kf.h[0] = *(const v8bf*)(kp1 + 32); kf.h[1] = *(const v8bf*)(kp1 + 40);
        sc1 = WMMA_BF16(qa1.v, kf.v, sc1);

        // ---- online softmax over the 32 keys (row = 8*half + r, col = nn)
#pragma unroll
        for (int r = 0; r < 8; ++r) {
            float s0 = sc0[r] * scale, s1 = sc1[r] * scale;
            float mx = red_max16(fmaxf(s0, s1));            // DPP16 reduction
            float mnew = fmaxf(mrow[r], mx);
            float ef = __expf(mrow[r] - mnew);
            float p0 = __expf(s0 - mnew), p1 = __expf(s1 - mnew);
            float rs = red_sum16(p0 + p1);                  // DPP16 reduction
            lrow[r] = lrow[r] * ef + rs;
            mrow[r] = mnew;
#pragma unroll
            for (int t = 0; t < 4; ++t) accv[t][r] *= ef;
            // stash P (C-layout) into LDS for re-shape into A-fragment layout
            ptile[w][8 * half + r][nn]      = (bf16_t)p0;
            ptile[w][8 * half + r][16 + nn] = (bf16_t)p1;
        }

        // ---- read P back as a 16x32 A-fragment (same-wave LDS: in-order)
        V16U pa;
        const bf16_t* pr = &ptile[w][nn][0];
        pa.h[0] = *(const v8bf*)(pr + 16 * half);
        pa.h[1] = *(const v8bf*)(pr + 16 * half + 8);

        // ---- O += P x V : Vt rows are contiguous in s -> direct B fragments
#pragma unroll
        for (int t = 0; t < 4; ++t) {
            const bf16_t* vp = Vt + (size_t)(bh * 64 + t * 16 + nn) * S + j + 16 * half;
            V16U vf;
            vf.h[0] = *(const v8bf*)vp;
            vf.h[1] = *(const v8bf*)(vp + 8);
            accv[t] = WMMA_BF16(pa.v, vf.v, accv[t]);
        }
    }

    // ---- normalize and store context in [B,S,H*64] layout
#pragma unroll
    for (int r = 0; r < 8; ++r) {
        float inv = 1.0f / lrow[r];
        int row = b * S + qt * 16 + 8 * half + r;
#pragma unroll
        for (int t = 0; t < 4; ++t)
            O[(size_t)row * D + h * 64 + t * 16 + nn] = (bf16_t)(accv[t][r] * inv);
    }
}

// ---------------------------------------------------------------------------
// out = w1*a + w2*b + res   (fp32; w1/w2 are 1-elem device arrays)
// ---------------------------------------------------------------------------
__global__ void combine_kernel(const float* __restrict__ a, const float* __restrict__ bb,
                               const float* __restrict__ res,
                               const float* __restrict__ w1, const float* __restrict__ w2,
                               float* __restrict__ out, int n) {
    int i = blockIdx.x * 256 + threadIdx.x;
    if (i >= n) return;
    out[i] = w1[0] * a[i] + w2[0] * bb[i] + res[i];
}

__global__ void add_inplace_kernel(float* __restrict__ out, const float* __restrict__ src, int n) {
    int i = blockIdx.x * 256 + threadIdx.x;
    if (i >= n) return;
    out[i] += src[i];
}

// ---------------------------------------------------------------------------
// Host orchestration
// ---------------------------------------------------------------------------
extern "C" void kernel_launch(void* const* d_in, const int* in_sizes, int n_in,
                              void* d_out, int out_size, void* d_ws, size_t ws_size,
                              hipStream_t stream) {
    (void)in_sizes; (void)n_in; (void)out_size; (void)ws_size;
    const int S = 1024, D = 768, F = 3072, NH = 12;
    const int Mrows = 4 * S;                 // 4096
    const size_t Aelem = (size_t)Mrows * D;  // 3,145,728

    auto in = [&](int i) { return (const float*)d_in[i]; };
    const float* x = in(0);      const float* y = in(1);
    const float* ln1x_g = in(2); const float* ln1x_b = in(3);
    const float* ln1y_g = in(4); const float* ln1y_b = in(5);
    const float* Wq = in(6);  const float* bq = in(7);
    const float* Wk = in(8);  const float* bk = in(9);
    const float* Wv = in(10); const float* bv = in(11);
    const float* Wo = in(12); const float* bo = in(13);
    const float* Wqd = in(14); const float* bqd = in(15);
    const float* Wkd = in(16); const float* bkd = in(17);
    const float* Wvd = in(18); const float* bvd = in(19);
    const float* Wod = in(20); const float* bod = in(21);
    const float* w11 = in(22); const float* w12 = in(23);
    const float* w21 = in(24); const float* w22 = in(25);
    const float* ln2x_g = in(26); const float* ln2x_b = in(27);
    const float* ln2y_g = in(28); const float* ln2y_b = in(29);
    const float* W1 = in(30);  const float* b1 = in(31);
    const float* W2 = in(32);  const float* b2 = in(33);
    const float* W1d = in(34); const float* b1d = in(35);
    const float* W2d = in(36); const float* b2d = in(37);

    // bump allocator over workspace
    char* ws = (char*)d_ws;
    size_t off = 0;
    auto alloc = [&](size_t bytes) -> char* {
        char* p = ws + off;
        off = (off + bytes + 255) & ~(size_t)255;
        return p;
    };

    bf16_t* wtq  = (bf16_t*)alloc((size_t)D * D * 2);
    bf16_t* wtk  = (bf16_t*)alloc((size_t)D * D * 2);
    bf16_t* wtv  = (bf16_t*)alloc((size_t)D * D * 2);
    bf16_t* wto  = (bf16_t*)alloc((size_t)D * D * 2);
    bf16_t* wtqd = (bf16_t*)alloc((size_t)D * D * 2);
    bf16_t* wtkd = (bf16_t*)alloc((size_t)D * D * 2);
    bf16_t* wtvd = (bf16_t*)alloc((size_t)D * D * 2);
    bf16_t* wtod = (bf16_t*)alloc((size_t)D * D * 2);
    bf16_t* wt1  = (bf16_t*)alloc((size_t)D * F * 2);
    bf16_t* wt2  = (bf16_t*)alloc((size_t)D * F * 2);
    bf16_t* wt1d = (bf16_t*)alloc((size_t)D * F * 2);
    bf16_t* wt2d = (bf16_t*)alloc((size_t)D * F * 2);
    bf16_t* xn   = (bf16_t*)alloc(Aelem * 2);
    bf16_t* yn   = (bf16_t*)alloc(Aelem * 2);
    bf16_t* qb   = (bf16_t*)alloc(Aelem * 2);
    bf16_t* kb   = (bf16_t*)alloc(Aelem * 2);
    bf16_t* vb   = (bf16_t*)alloc(Aelem * 2);
    bf16_t* qdb  = (bf16_t*)alloc(Aelem * 2);
    bf16_t* kdb  = (bf16_t*)alloc(Aelem * 2);
    bf16_t* vdb  = (bf16_t*)alloc(Aelem * 2);
    bf16_t* vtb  = (bf16_t*)alloc(Aelem * 2);
    bf16_t* vdtb = (bf16_t*)alloc(Aelem * 2);
    bf16_t* ctx  = (bf16_t*)alloc(Aelem * 2);
    bf16_t* hbuf = (bf16_t*)alloc((size_t)Mrows * F * 2);
    float*  asx  = (float*)alloc(Aelem * 4);
    float*  acx  = (float*)alloc(Aelem * 4);
    float*  asy  = (float*)alloc(Aelem * 4);
    float*  acy  = (float*)alloc(Aelem * 4);
    float*  mlpb = (float*)alloc(Aelem * 4);

    float* outx = (float*)d_out;
    float* outy = outx + Aelem;

    const int ew = (D * D + 255) / 256;
    const int ef = (D * F + 255) / 256;
    const int ea = (int)((Aelem + 255) / 256);

    // 1. weights -> bf16, transposed to [N][K]
    wtrans_kernel<<<ew, 256, 0, stream>>>(Wq, wtq, D, D);
    wtrans_kernel<<<ew, 256, 0, stream>>>(Wk, wtk, D, D);
    wtrans_kernel<<<ew, 256, 0, stream>>>(Wv, wtv, D, D);
    wtrans_kernel<<<ew, 256, 0, stream>>>(Wo, wto, D, D);
    wtrans_kernel<<<ew, 256, 0, stream>>>(Wqd, wtqd, D, D);
    wtrans_kernel<<<ew, 256, 0, stream>>>(Wkd, wtkd, D, D);
    wtrans_kernel<<<ew, 256, 0, stream>>>(Wvd, wtvd, D, D);
    wtrans_kernel<<<ew, 256, 0, stream>>>(Wod, wtod, D, D);
    wtrans_kernel<<<ef, 256, 0, stream>>>(W1,  wt1,  D, F);
    wtrans_kernel<<<ef, 256, 0, stream>>>(W2,  wt2,  F, D);
    wtrans_kernel<<<ef, 256, 0, stream>>>(W1d, wt1d, D, F);
    wtrans_kernel<<<ef, 256, 0, stream>>>(W2d, wt2d, F, D);

    // 2. LN1
    ln_bf16_kernel<<<Mrows, 256, 0, stream>>>(x, ln1x_g, ln1x_b, xn);
    ln_bf16_kernel<<<Mrows, 256, 0, stream>>>(y, ln1y_g, ln1y_b, yn);

    // 3. QKV projections (bf16 out); block tile 128x128
    dim3 gD(D / 128, Mrows / 128);           // (6, 32)
    gemm_bf16_kernel<true, false><<<gD, 256, 0, stream>>>(xn, wtq, bq, qb, Mrows, D, D);
    gemm_bf16_kernel<true, false><<<gD, 256, 0, stream>>>(xn, wtk, bk, kb, Mrows, D, D);
    gemm_bf16_kernel<true, false><<<gD, 256, 0, stream>>>(xn, wtv, bv, vb, Mrows, D, D);
    gemm_bf16_kernel<true, false><<<gD, 256, 0, stream>>>(yn, wtqd, bqd, qdb, Mrows, D, D);
    gemm_bf16_kernel<true, false><<<gD, 256, 0, stream>>>(yn, wtkd, bkd, kdb, Mrows, D, D);
    gemm_bf16_kernel<true, false><<<gD, 256, 0, stream>>>(yn, wtvd, bvd, vdb, Mrows, D, D);

    // 4. V transposes for the PV WMMA B-fragments
    vtrans_kernel<<<ea, 256, 0, stream>>>(vb,  vtb);
    vtrans_kernel<<<ea, 256, 0, stream>>>(vdb, vdtb);

    // 5. four attentions + output projections (fp32 out)
    dim3 gA(S / 64, 4 * NH);                 // (16, 48)
    attn_kernel<<<gA, 128, 0, stream>>>(qb, kb, vtb, ctx);
    gemm_bf16_kernel<false, false><<<gD, 256, 0, stream>>>(ctx, wto, bo, asx, Mrows, D, D);
    attn_kernel<<<gA, 128, 0, stream>>>(qdb, kdb, vdtb, ctx);
    gemm_bf16_kernel<false, false><<<gD, 256, 0, stream>>>(ctx, wtod, bod, asy, Mrows, D, D);
    attn_kernel<<<gA, 128, 0, stream>>>(qb, kdb, vdtb, ctx);
    gemm_bf16_kernel<false, false><<<gD, 256, 0, stream>>>(ctx, wto, bo, acx, Mrows, D, D);
    attn_kernel<<<gA, 128, 0, stream>>>(qdb, kb, vtb, ctx);
    gemm_bf16_kernel<false, false><<<gD, 256, 0, stream>>>(ctx, wtod, bod, acy, Mrows, D, D);

    // 6. weighted combine + residual -> d_out (mid activations)
    combine_kernel<<<ea, 256, 0, stream>>>(asx, acx, x, w11, w12, outx, (int)Aelem);
    combine_kernel<<<ea, 256, 0, stream>>>(asy, acy, y, w21, w22, outy, (int)Aelem);

    // 7. LN2 (reuse xn/yn)
    ln_bf16_kernel<<<Mrows, 256, 0, stream>>>(outx, ln2x_g, ln2x_b, xn);
    ln_bf16_kernel<<<Mrows, 256, 0, stream>>>(outy, ln2y_g, ln2y_b, yn);

    // 8. MLPs: GEMM+GELU (bf16 hidden) -> GEMM (fp32) -> residual add
    dim3 gF(F / 128, Mrows / 128);           // (24, 32)
    gemm_bf16_kernel<true, true><<<gF, 256, 0, stream>>>(xn, wt1, b1, hbuf, Mrows, F, D);
    gemm_bf16_kernel<false, false><<<gD, 256, 0, stream>>>(hbuf, wt2, b2, mlpb, Mrows, D, F);
    add_inplace_kernel<<<ea, 256, 0, stream>>>(outx, mlpb, (int)Aelem);

    gemm_bf16_kernel<true, true><<<gF, 256, 0, stream>>>(yn, wt1d, b1d, hbuf, Mrows, F, D);
    gemm_bf16_kernel<false, false><<<gD, 256, 0, stream>>>(hbuf, wt2d, b2d, mlpb, Mrows, D, F);
    add_inplace_kernel<<<ea, 256, 0, stream>>>(outy, mlpb, (int)Aelem);
}
